// Block_28114855919934
// MI455X (gfx1250) — matrix operand order repaired and checked by reading, not compiled, not run on hardware
//
#include <hip/hip_runtime.h>
#include <hip/hip_bf16.h>

// ---------------------------------------------------------------------------
// Types for CDNA5 WMMA (wave32): D(16x16 f32) = A(16x32 bf16) x B(32x16 bf16) + C
// ---------------------------------------------------------------------------
typedef __attribute__((ext_vector_type(16))) __bf16 v16bf;
typedef __attribute__((ext_vector_type(8)))  __bf16 v8bf;
typedef __attribute__((ext_vector_type(8)))  float  v8f;

__device__ inline v8f wmma_bf16(v16bf a, v16bf b, v8f c) {
    // (neg_a, A, neg_b, B, c_mod, C, reuse_a, reuse_b)
    return __builtin_amdgcn_wmma_f32_16x16x32_bf16(false, a, false, b, (short)0, c, false, false);
}

// 16 contiguous bf16 -> fragment (contiguous k order r[idx] = K kbase+idx; usable only if A and B both use it)
__device__ inline v16bf ld16(const __bf16* p) {
    v8bf lo = *(const v8bf*)(p);
    v8bf hi = *(const v8bf*)(p + 8);
    v16bf r;
#pragma unroll
    for (int i = 0; i < 8; ++i) { r[i] = lo[i]; r[8 + i] = hi[i]; }
    return r;
}
// 8 bf16 @ p plus 8 bf16 @ p+16 (A-operand: a[0..7]=K kh+0..7, a[8..15]=K 16+kh+0..7)
__device__ inline v16bf ld8gap(const __bf16* p) {
    v8bf lo = *(const v8bf*)(p);
    v8bf hi = *(const v8bf*)(p + 16);
    v16bf r;
#pragma unroll
    for (int i = 0; i < 8; ++i) { r[i] = lo[i]; r[8 + i] = hi[i]; }
    return r;
}

__device__ inline float gelu_tanh(float v) {
    const float c = 0.7978845608028654f; // sqrt(2/pi)
    float u = c * (v + 0.044715f * v * v * v);
    return 0.5f * v * (1.0f + tanhf(u));
}

// ---------------------------------------------------------------------------
// Weight convert + transpose: in f32 [K][N] row-major -> out bf16 [N][K]
// 32x32 tiles through LDS, both sides coalesced.
// ---------------------------------------------------------------------------
__global__ __launch_bounds__(256) void cvt_transpose(const float* __restrict__ in,
                                                     __bf16* __restrict__ out,
                                                     int K, int N) {
    __shared__ float tile[32][33];
    const int k0 = blockIdx.y * 32, n0 = blockIdx.x * 32;
    const int tx = threadIdx.x & 31, ty = threadIdx.x >> 5;
#pragma unroll
    for (int r = ty; r < 32; r += 8)
        tile[r][tx] = in[(size_t)(k0 + r) * N + n0 + tx];
    __syncthreads();
#pragma unroll
    for (int r = ty; r < 32; r += 8)
        out[(size_t)(n0 + r) * K + k0 + tx] = (__bf16)tile[tx][r];
}

// ---------------------------------------------------------------------------
// V transpose: qkv bf16 [B*T, 3C] (v at col 2048 + h*64 + d) -> vT [B][H][64][T]
// ---------------------------------------------------------------------------
__global__ __launch_bounds__(256) void v_transpose(const __bf16* __restrict__ qkv,
                                                   __bf16* __restrict__ vT) {
    const int idx = blockIdx.x * 256 + threadIdx.x; // over B*H*64*T = 4M
    const int t = idx & 2047;
    const int d = (idx >> 11) & 63;
    const int h = (idx >> 17) & 15;
    const int b = idx >> 21;
    vT[idx] = qkv[(size_t)(b * 2048 + t) * 3072 + 2048 + h * 64 + d];
}

// ---------------------------------------------------------------------------
// LayerNorm over C=1024, one row per block, bf16 output
// ---------------------------------------------------------------------------
__global__ __launch_bounds__(256) void layernorm_bf16(const float* __restrict__ x,
                                                      const float* __restrict__ g,
                                                      const float* __restrict__ bb,
                                                      __bf16* __restrict__ out, int C_) {
    __shared__ float red[16];
    const int row = blockIdx.x;
    const float* xr = x + (size_t)row * C_;
    float s = 0.f, s2 = 0.f;
    for (int i = threadIdx.x; i < C_; i += 256) {
        float v = xr[i];
        s += v; s2 += v * v;
    }
    for (int off = 16; off; off >>= 1) {
        s  += __shfl_xor(s, off, 32);
        s2 += __shfl_xor(s2, off, 32);
    }
    const int wave = threadIdx.x >> 5, lane = threadIdx.x & 31;
    if (lane == 0) { red[wave] = s; red[8 + wave] = s2; }
    __syncthreads();
    s = 0.f; s2 = 0.f;
#pragma unroll
    for (int w = 0; w < 8; ++w) { s += red[w]; s2 += red[8 + w]; }
    const float mu = s / (float)C_;
    const float var = s2 / (float)C_ - mu * mu;
    const float rs = rsqrtf(var + 1e-5f);
    for (int i = threadIdx.x; i < C_; i += 256) {
        float v = (xr[i] - mu) * rs * g[i] + bb[i];
        out[(size_t)row * C_ + i] = (__bf16)v;
    }
}

// ---------------------------------------------------------------------------
// WMMA GEMM, fragment-direct from global (weights pre-transposed, all operands
// contiguous per lane -> b128 loads; 192MB L2 provides reuse). No LDS/barriers.
// out[M,N] = A[M,K](bf16) @ WT[N,K](bf16)^T + bias, optional GELU / residual.
// 256 threads = 8 waves; wave tile 32x64 (2 A-frags x 4 B-frags, 8 WMMA/K-step);
// block tile 128x128. K-tile two steps ahead (k0+64) prefetched via global_prefetch.
// ---------------------------------------------------------------------------
template <bool GELU, bool RESID, bool OUTBF>
__global__ __launch_bounds__(256) void gemm_wmma(const __bf16* __restrict__ A,
                                                 const __bf16* __restrict__ WT,
                                                 const float* __restrict__ bias,
                                                 const float* __restrict__ resid,
                                                 float* __restrict__ outf,
                                                 __bf16* __restrict__ outb,
                                                 int M, int N, int K) {
    const int wave = threadIdx.x >> 5;
    const int lane = threadIdx.x & 31;
    const int m0 = blockIdx.y * 128 + (wave >> 1) * 32;
    const int n0 = blockIdx.x * 128 + (wave & 1) * 64;
    const int lm = lane & 15;
    const int kh  = (lane >> 4) << 3;  // A half: 0 or 8
    const int kb16 = (lane >> 4) << 3; // B half: 0 or 8

    v8f acc[2][4] = {};

    const __bf16* a0p = A + (size_t)(m0 + lm) * K + kh;
    const __bf16* a1p = A + (size_t)(m0 + 16 + lm) * K + kh;
    const __bf16* b0p = WT + (size_t)(n0 + 0  + lm) * K + kb16;
    const __bf16* b1p = WT + (size_t)(n0 + 16 + lm) * K + kb16;
    const __bf16* b2p = WT + (size_t)(n0 + 32 + lm) * K + kb16;
    const __bf16* b3p = WT + (size_t)(n0 + 48 + lm) * K + kb16;

    for (int k0 = 0; k0 < K; k0 += 32) {
        if (k0 + 64 <= K - 32) { // uniform branch: prefetch K-tile two steps ahead (global_prefetch_b8)
            __builtin_prefetch(a0p + k0 + 64, 0, 0);
            __builtin_prefetch(a1p + k0 + 64, 0, 0);
            __builtin_prefetch(b0p + k0 + 64, 0, 0);
            __builtin_prefetch(b1p + k0 + 64, 0, 0);
            __builtin_prefetch(b2p + k0 + 64, 0, 0);
            __builtin_prefetch(b3p + k0 + 64, 0, 0);
        }
        const v16bf a0 = ld8gap(a0p + k0);
        const v16bf a1 = ld8gap(a1p + k0);
        {
            const v16bf b = ld8gap(b0p + k0);
            acc[0][0] = wmma_bf16(a0, b, acc[0][0]);
            acc[1][0] = wmma_bf16(a1, b, acc[1][0]);
        }
        {
            const v16bf b = ld8gap(b1p + k0);
            acc[0][1] = wmma_bf16(a0, b, acc[0][1]);
            acc[1][1] = wmma_bf16(a1, b, acc[1][1]);
        }
        {
            const v16bf b = ld8gap(b2p + k0);
            acc[0][2] = wmma_bf16(a0, b, acc[0][2]);
            acc[1][2] = wmma_bf16(a1, b, acc[1][2]);
        }
        {
            const v16bf b = ld8gap(b3p + k0);
            acc[0][3] = wmma_bf16(a0, b, acc[0][3]);
            acc[1][3] = wmma_bf16(a1, b, acc[1][3]);
        }
    }

    // Epilogue. C/D layout: lanes 0-15 -> N=lane, M=r; lanes 16-31 -> N=lane-16, M=r+8.
#pragma unroll
    for (int hmf = 0; hmf < 2; ++hmf) {
#pragma unroll
        for (int f = 0; f < 4; ++f) {
#pragma unroll
            for (int r = 0; r < 8; ++r) {
                const int m = m0 + hmf * 16 + r + ((lane >> 4) << 3);
                const int n = n0 + f * 16 + (lane & 15);
                float v = acc[hmf][f][r] + bias[n];
                if (GELU) v = gelu_tanh(v);
                if (RESID) v += resid[(size_t)m * N + n];
                if (OUTBF) outb[(size_t)m * N + n] = (__bf16)v;
                else       outf[(size_t)m * N + n] = v;
            }
        }
    }
}

// ---------------------------------------------------------------------------
// Flash attention. qkv: bf16 [B*T, 3C] (q|k|v, C=1024 = 16 heads x 64),
// vT: bf16 [B][H][64][T]. One wave per 16-query block of one (b,h); key
// blocks of 32 with online softmax; WMMA for q@k^T and p@v. 1/sqrt(HD) is
// folded into the Q fragment (exact in bf16). Causal masking only on the
// wave-uniform straddling key blocks. No block-wide barriers (divergent
// causal trip counts) -> per-wave LDS + s_wait_dscnt.
// ---------------------------------------------------------------------------
__global__ __launch_bounds__(256) void attn_flash(const __bf16* __restrict__ qkv,
                                                  const __bf16* __restrict__ vT,
                                                  __bf16* __restrict__ yout) {
    __shared__ alignas(16) __bf16 Plds[8][16 * 32];
    const int lane = threadIdx.x & 31;
    const int wave = threadIdx.x >> 5;
    const int wid = blockIdx.x * 8 + wave;

    const int qb = wid & 127;          // T/16 = 128 query blocks
    const int h  = (wid >> 7) & 15;    // 16 heads
    const int b  = wid >> 11;          // batch
    const size_t RS = 3072;            // qkv row stride
    const __bf16* base = qkv + (size_t)b * 2048 * RS;
    const __bf16* vTb  = vT + ((size_t)(b * 16 + h) * 64) * 2048;

    const int nloc = lane & 15;
    const int half8  = (lane >> 4) << 3;  // 0 or 8
    const int half16 = (lane >> 4) << 4;  // 0 or 16

    // Q fragments (d = 0..31 and 32..63), pre-scaled by 1/sqrt(64) = 0.125
    v16bf qf[2];
    {
        const int qrow = qb * 16 + nloc;
#pragma unroll
        for (int df = 0; df < 2; ++df) {
            qf[df] = ld8gap(base + (size_t)qrow * RS + h * 64 + df * 32 + half8);
#pragma unroll
            for (int i = 0; i < 16; ++i)
                qf[df][i] = (__bf16)((float)qf[df][i] * 0.125f);
        }
    }

    float mstate[8], lstate[8];
    v8f o[4] = {};
#pragma unroll
    for (int r = 0; r < 8; ++r) { mstate[r] = -__builtin_inff(); lstate[r] = 0.f; }

    __bf16* Pw = &Plds[wave][0];
    const int nkb = ((qb * 16 + 15) >> 5) + 1; // causal: key blocks of 32

    for (int kb = 0; kb < nkb; ++kb) {
        // ---- S = (Q @ K^T) for 32 keys: two 16x16 accumulators
        v8f s0 = {}, s1 = {};
        const int key0 = kb * 32 + nloc;
#pragma unroll
        for (int df = 0; df < 2; ++df) {
            const v16bf kf0 = ld8gap(base + (size_t)key0 * RS        + 1024 + h * 64 + df * 32 + half8);
            const v16bf kf1 = ld8gap(base + (size_t)(key0 + 16) * RS + 1024 + h * 64 + df * 32 + half8);
            s0 = wmma_bf16(qf[df], kf0, s0);
            s1 = wmma_bf16(qf[df], kf1, s1);
        }

        // ---- online softmax; masking only on straddling blocks (wave-uniform)
        const bool needMask = (kb * 32 + 31) > (qb * 16); // qb,kb wave-uniform -> SCC branch
#pragma unroll
        for (int r = 0; r < 8; ++r) {
            float v0 = s0[r];
            float v1 = s1[r];
            if (needMask) {
                const int m = qb * 16 + r + half8;
                v0 = (kb * 32 + nloc > m)      ? -__builtin_inff() : v0;
                v1 = (kb * 32 + 16 + nloc > m) ? -__builtin_inff() : v1;
            }
            float mx = fmaxf(v0, v1);
            for (int off = 1; off < 16; off <<= 1) mx = fmaxf(mx, __shfl_xor(mx, off, 32));
            const float mnew  = fmaxf(mstate[r], mx);
            const float alpha = __expf(mstate[r] - mnew);
            const float p0 = __expf(v0 - mnew);
            const float p1 = __expf(v1 - mnew);
            float ls = p0 + p1;
            for (int off = 1; off < 16; off <<= 1) ls += __shfl_xor(ls, off, 32);
            lstate[r] = lstate[r] * alpha + ls;
            mstate[r] = mnew;
#pragma unroll
            for (int f = 0; f < 4; ++f) o[f][r] *= alpha;

            const int mloc = r + half8;
            Pw[mloc * 32 + nloc]      = (__bf16)p0;
            Pw[mloc * 32 + nloc + 16] = (__bf16)p1;
        }
        asm volatile("s_wait_dscnt 0" ::: "memory");

        // ---- O += P @ V  (P from per-wave LDS in A layout; V from vT, contiguous)
        const v16bf pf = ld8gap(Pw + nloc * 32 + half8);
#pragma unroll
        for (int f = 0; f < 4; ++f) {
            const int d = f * 16 + nloc;
            const v16bf vf = ld8gap(vTb + (size_t)d * 2048 + kb * 32 + half8);
            o[f] = wmma_bf16(pf, vf, o[f]);
        }
        asm volatile("s_wait_dscnt 0" ::: "memory");
    }

    // ---- write y (merge heads): y[b*T + t, h*64 + d], bf16
#pragma unroll
    for (int f = 0; f < 4; ++f) {
#pragma unroll
        for (int r = 0; r < 8; ++r) {
            const int tq = qb * 16 + r + half8;
            const float val = o[f][r] / lstate[r];
            yout[((size_t)(b * 2048 + tq)) * 1024 + h * 64 + f * 16 + nloc] = (__bf16)val;
        }
    }
}

// ---------------------------------------------------------------------------
// Launch
// ---------------------------------------------------------------------------
extern "C" void kernel_launch(void* const* d_in, const int* in_sizes, int n_in,
                              void* d_out, int out_size, void* d_ws, size_t ws_size,
                              hipStream_t stream) {
    (void)in_sizes; (void)n_in; (void)out_size; (void)ws_size;
    const float* x      = (const float*)d_in[0];
    const float* ln1_g  = (const float*)d_in[1];
    const float* ln1_b  = (const float*)d_in[2];
    const float* W_attn = (const float*)d_in[3];
    const float* b_attn = (const float*)d_in[4];
    const float* W_proj = (const float*)d_in[5];
    const float* b_proj = (const float*)d_in[6];
    const float* ln2_g  = (const float*)d_in[7];
    const float* ln2_b  = (const float*)d_in[8];
    const float* W_fc   = (const float*)d_in[9];
    const float* b_fc   = (const float*)d_in[10];
    const float* W_fc2  = (const float*)d_in[11];
    const float* b_fc2  = (const float*)d_in[12];
    float* out = (float*)d_out;

    const int BT = 2 * 2048;   // 4096 rows
    const int C  = 1024;

    char* ws = (char*)d_ws;
    size_t off = 0;
    auto carve = [&](size_t bytes) -> void* {
        void* p = ws + off;
        off = (off + bytes + 255) & ~(size_t)255;
        return p;
    };
    __bf16* WattnT = (__bf16*)carve((size_t)C * 3 * C * 2);   // [3C][C]
    __bf16* WprojT = (__bf16*)carve((size_t)C * C * 2);       // [C][C]
    __bf16* WfcT   = (__bf16*)carve((size_t)C * 4 * C * 2);   // [4C][C]
    __bf16* Wfc2T  = (__bf16*)carve((size_t)4 * C * C * 2);   // [C][4C]
    __bf16* h_b    = (__bf16*)carve((size_t)BT * C * 2);
    __bf16* qkv_b  = (__bf16*)carve((size_t)BT * 3 * C * 2);
    __bf16* vT_b   = (__bf16*)carve((size_t)BT * C * 2);      // [B][H][64][T]
    __bf16* y_b    = (__bf16*)carve((size_t)BT * C * 2);
    float*  x1     = (float*)carve((size_t)BT * C * 4);
    __bf16* h2_b   = (__bf16*)carve((size_t)BT * C * 2);
    __bf16* a_b    = (__bf16*)carve((size_t)BT * 4 * C * 2);

    // Weight convert + transpose (f32 [K][N] -> bf16 [N][K])
    cvt_transpose<<<dim3(3072 / 32, 1024 / 32), 256, 0, stream>>>(W_attn, WattnT, 1024, 3072);
    cvt_transpose<<<dim3(1024 / 32, 1024 / 32), 256, 0, stream>>>(W_proj, WprojT, 1024, 1024);
    cvt_transpose<<<dim3(4096 / 32, 1024 / 32), 256, 0, stream>>>(W_fc,   WfcT,   1024, 4096);
    cvt_transpose<<<dim3(1024 / 32, 4096 / 32), 256, 0, stream>>>(W_fc2,  Wfc2T,  4096, 1024);

    // LN1 -> h
    layernorm_bf16<<<BT, 256, 0, stream>>>(x, ln1_g, ln1_b, h_b, C);
    // qkv = h @ W_attn + b_attn  (bf16 out)  [4096 x 3072]
    gemm_wmma<false, false, true><<<dim3(3072 / 128, BT / 128), 256, 0, stream>>>(
        h_b, WattnT, b_attn, nullptr, nullptr, qkv_b, BT, 3 * C, C);
    // v transpose for p@v fragments
    v_transpose<<<(BT * C) / 256, 256, 0, stream>>>(qkv_b, vT_b);
    // attention -> y (bf16, heads merged)
    attn_flash<<<512, 256, 0, stream>>>(qkv_b, vT_b, y_b);
    // x1 = x + y @ W_proj + b_proj  (f32 out)
    gemm_wmma<false, true, false><<<dim3(1024 / 128, BT / 128), 256, 0, stream>>>(
        y_b, WprojT, b_proj, x, x1, nullptr, BT, C, C);
    // LN2 -> h2
    layernorm_bf16<<<BT, 256, 0, stream>>>(x1, ln2_g, ln2_b, h2_b, C);
    // a = gelu(h2 @ W_fc + b_fc)  (bf16 out)  [4096 x 4096]
    gemm_wmma<true, false, true><<<dim3(4096 / 128, BT / 128), 256, 0, stream>>>(
        h2_b, WfcT, b_fc, nullptr, nullptr, a_b, BT, 4 * C, C);
    // out = x1 + a @ W_fc2 + b_fc2  (f32 out)
    gemm_wmma<false, true, false><<<dim3(1024 / 128, BT / 128), 256, 0, stream>>>(
        a_b, Wfc2T, b_fc2, x1, out, nullptr, BT, C, 4 * C);
}
